// VIMAttention_9156870275464
// MI455X (gfx1250) — matrix operand order repaired and checked by reading, not compiled, not run on hardware
//
#include <hip/hip_runtime.h>
#include <hip/hip_bf16.h>
#include <math.h>

#define D_MODEL 256
#define D_INNER 512
#define D_STATE 16
#define DT_RANK 16
#define DT_PAD  32            // dt GEMM K padded to one WMMA K-step
#define D_CONV  4
#define N_LAYERS 8
#define L1      768
#define SEQ     1536
#define BSZ     4
#define M_ROWS  (BSZ * SEQ)   // 6144
#define XDBL_W  (DT_RANK + 2 * D_STATE)  // 48

typedef __attribute__((ext_vector_type(8)))  _Float16 v8h;
typedef __attribute__((ext_vector_type(16))) _Float16 v16h;
typedef __attribute__((ext_vector_type(8)))  float    v8f;

// ---------------------------------------------------------------------------
// Generic f32 -> f16 conversion (weights)
// ---------------------------------------------------------------------------
__global__ void cvt_f16_kernel(const float* __restrict__ src,
                               _Float16* __restrict__ dst, int n) {
  int idx = blockIdx.x * blockDim.x + threadIdx.x;
  if (idx >= n) return;
  dst[idx] = (_Float16)src[idx];
}

// f32 (rows x 16) -> f16 (rows x 32), upper 16 zero-padded (dt_proj weights)
__global__ void cvt_pad_kernel(const float* __restrict__ src,
                               _Float16* __restrict__ dst, int n) {
  int idx = blockIdx.x * blockDim.x + threadIdx.x;
  if (idx >= n) return;
  int r = idx >> 5, c = idx & 31;
  dst[idx] = (c < DT_RANK) ? (_Float16)src[(size_t)r * DT_RANK + c]
                           : (_Float16)0.f;
}

// ---------------------------------------------------------------------------
// Pack: hidden[b,l,d] = concat(rgb, ir)(b,l,d) + pos_embed[l,d]
// ---------------------------------------------------------------------------
__global__ void pack_kernel(const float* __restrict__ rgb,
                            const float* __restrict__ ir,
                            const float* __restrict__ pos,
                            float* __restrict__ hid, int n) {
  int idx = blockIdx.x * blockDim.x + threadIdx.x;
  if (idx >= n) return;
  int d = idx & (D_MODEL - 1);
  int r = idx >> 8;
  int b = r / SEQ, l = r % SEQ;
  float v = pos[(size_t)l * D_MODEL + d];
  if (l < L1) v += rgb[((size_t)b * D_MODEL + d) * L1 + l];
  else        v += ir [((size_t)b * D_MODEL + d) * L1 + (l - L1)];
  hid[idx] = v;
}

// ---------------------------------------------------------------------------
// Fused add + LayerNorm (prenorm). One wave32 per 256-wide token row.
// ---------------------------------------------------------------------------
__global__ __launch_bounds__(256) void addnorm_kernel(
    const float* __restrict__ hid, const float* __restrict__ resin,
    const float* __restrict__ w, const float* __restrict__ bias,
    float* __restrict__ res_out, _Float16* __restrict__ nh,
    float* __restrict__ nf, int flip) {
  const int lane = threadIdx.x & 31;
  const int wave = threadIdx.x >> 5;
  const int row  = blockIdx.x * 8 + wave;
  const int b    = row / SEQ, l = row % SEQ;
  const int rr   = flip ? (b * SEQ + (SEQ - 1 - l)) : row;
  const int d0   = lane * 8;

  float v[8];
  {
    const float* hp = hid + (size_t)rr * D_MODEL + d0;
    float4 a = ((const float4*)hp)[0];
    float4 c = ((const float4*)hp)[1];
    v[0] = a.x; v[1] = a.y; v[2] = a.z; v[3] = a.w;
    v[4] = c.x; v[5] = c.y; v[6] = c.z; v[7] = c.w;
  }
  if (resin) {
    const float* rp = resin + (size_t)rr * D_MODEL + d0;
    float4 a = ((const float4*)rp)[0];
    float4 c = ((const float4*)rp)[1];
    v[0] += a.x; v[1] += a.y; v[2] += a.z; v[3] += a.w;
    v[4] += c.x; v[5] += c.y; v[6] += c.z; v[7] += c.w;
  }
  {
    float* op = res_out + (size_t)row * D_MODEL + d0;
    ((float4*)op)[0] = make_float4(v[0], v[1], v[2], v[3]);
    ((float4*)op)[1] = make_float4(v[4], v[5], v[6], v[7]);
  }
  float s = 0.f;
  #pragma unroll
  for (int j = 0; j < 8; ++j) s += v[j];
  #pragma unroll
  for (int off = 16; off > 0; off >>= 1) s += __shfl_xor(s, off, 32);
  const float mu = s * (1.f / D_MODEL);
  float vs = 0.f;
  #pragma unroll
  for (int j = 0; j < 8; ++j) { float t = v[j] - mu; vs += t * t; }
  #pragma unroll
  for (int off = 16; off > 0; off >>= 1) vs += __shfl_xor(vs, off, 32);
  const float inv = rsqrtf(vs * (1.f / D_MODEL) + 1e-5f);

  if (nh) {
    v8h o;
    #pragma unroll
    for (int j = 0; j < 8; ++j)
      o[j] = (_Float16)((v[j] - mu) * inv * w[d0 + j] + bias[d0 + j]);
    *(v8h*)(nh + (size_t)row * D_MODEL + d0) = o;
  }
  if (nf) {
    float o[8];
    #pragma unroll
    for (int j = 0; j < 8; ++j)
      o[j] = (v[j] - mu) * inv * w[d0 + j] + bias[d0 + j];
    float* op = nf + (size_t)row * D_MODEL + d0;
    ((float4*)op)[0] = make_float4(o[0], o[1], o[2], o[3]);
    ((float4*)op)[1] = make_float4(o[4], o[5], o[6], o[7]);
  }
}

// ---------------------------------------------------------------------------
// WMMA GEMM: C[m,n] = sum_k A[m,k] * W[n,k]
//   A: (M_ROWS x K) f16 row-major, W: (N x K) f16 row-major, C f32.
//   K compile-time (multiple of 32): no guards, fully unrolled.
//   4 waves/block (one N tile each) x 4 M tiles/wave; the K loop is
//   software-pipelined by one step so each WMMA group only waits on the
//   previous prefetch group (partial loadcnt waits, loads stay in flight).
// ---------------------------------------------------------------------------
template <int K>
__global__ __launch_bounds__(128) void gemm_f16_wmma(
    const _Float16* __restrict__ A, const _Float16* __restrict__ W,
    float* __restrict__ C, int N) {
  const int lane = threadIdx.x & 31;
  const int wave = threadIdx.x >> 5;
  const int tn = blockIdx.x * 4 + wave;
  if (tn * 16 >= N) return;                 // wave-uniform; no barriers used
  const int g16 = lane >> 4;                // which half of the wave
  const int i16 = lane & 15;
  const int tm0 = blockIdx.y * 4;

  // B fragment: lane holds column (tn*16+i16), two contiguous 8-half runs at g16*8 and g16*8+16
  const _Float16* Wp = W + (size_t)(tn * 16 + i16) * K + g16 * 8;
  // A fragment: lane holds row, two contiguous 8-half runs at kA0 and kA0+16
  const _Float16* Ap0 = A + (size_t)((tm0 + 0) * 16 + i16) * K + g16 * 8;
  const _Float16* Ap1 = A + (size_t)((tm0 + 1) * 16 + i16) * K + g16 * 8;
  const _Float16* Ap2 = A + (size_t)((tm0 + 2) * 16 + i16) * K + g16 * 8;
  const _Float16* Ap3 = A + (size_t)((tm0 + 3) * 16 + i16) * K + g16 * 8;

#define LOAD_A(P, KK)                                                         \
  __builtin_shufflevector(*(const v8h*)((P) + (KK)),                          \
                          *(const v8h*)((P) + (KK) + 16),                     \
                          0, 1, 2, 3, 4, 5, 6, 7,                             \
                          8, 9, 10, 11, 12, 13, 14, 15)

  v8f acc0 = {}, acc1 = {}, acc2 = {}, acc3 = {};

  // Prologue: fetch k-step 0.
  v16h bCur = LOAD_A(Wp, 0);
  v16h a0 = LOAD_A(Ap0, 0);
  v16h a1 = LOAD_A(Ap1, 0);
  v16h a2 = LOAD_A(Ap2, 0);
  v16h a3 = LOAD_A(Ap3, 0);

  #pragma unroll
  for (int k0 = 0; k0 < K; k0 += 32) {
    v16h bNxt, n0, n1, n2, n3;
    if (k0 + 32 < K) {                      // static after full unroll
      bNxt = LOAD_A(Wp, k0 + 32);
      n0 = LOAD_A(Ap0, k0 + 32);
      n1 = LOAD_A(Ap1, k0 + 32);
      n2 = LOAD_A(Ap2, k0 + 32);
      n3 = LOAD_A(Ap3, k0 + 32);
    }
    acc0 = __builtin_amdgcn_wmma_f32_16x16x32_f16(false, a0, false, bCur,
                                                  (short)0, acc0, false, false);
    acc1 = __builtin_amdgcn_wmma_f32_16x16x32_f16(false, a1, false, bCur,
                                                  (short)0, acc1, false, false);
    acc2 = __builtin_amdgcn_wmma_f32_16x16x32_f16(false, a2, false, bCur,
                                                  (short)0, acc2, false, false);
    acc3 = __builtin_amdgcn_wmma_f32_16x16x32_f16(false, a3, false, bCur,
                                                  (short)0, acc3, false, false);
    if (k0 + 32 < K) {
      bCur = bNxt; a0 = n0; a1 = n1; a2 = n2; a3 = n3;
    }
  }
#undef LOAD_A

  const int cn = tn * 16 + i16;
  {
    float* Cp = C + (size_t)((tm0 + 0) * 16 + g16 * 8) * N + cn;
    #pragma unroll
    for (int r = 0; r < 8; ++r) Cp[(size_t)r * N] = acc0[r];
  }
  {
    float* Cp = C + (size_t)((tm0 + 1) * 16 + g16 * 8) * N + cn;
    #pragma unroll
    for (int r = 0; r < 8; ++r) Cp[(size_t)r * N] = acc1[r];
  }
  {
    float* Cp = C + (size_t)((tm0 + 2) * 16 + g16 * 8) * N + cn;
    #pragma unroll
    for (int r = 0; r < 8; ++r) Cp[(size_t)r * N] = acc2[r];
  }
  {
    float* Cp = C + (size_t)((tm0 + 3) * 16 + g16 * 8) * N + cn;
    #pragma unroll
    for (int r = 0; r < 8; ++r) Cp[(size_t)r * N] = acc3[r];
  }
}

// ---------------------------------------------------------------------------
// Causal depthwise conv1d (k=4) + bias + SiLU. Reads x half of xz.
// ---------------------------------------------------------------------------
__global__ void conv_silu_kernel(const float* __restrict__ xz,
                                 const float* __restrict__ cw,
                                 const float* __restrict__ cb,
                                 float* __restrict__ xf,
                                 _Float16* __restrict__ xh, int n) {
  int idx = blockIdx.x * blockDim.x + threadIdx.x;
  if (idx >= n) return;
  int d = idx & (D_INNER - 1);
  int r = idx >> 9;
  int b = r / SEQ, l = r % SEQ;
  float acc = cb[d];
  #pragma unroll
  for (int j = 0; j < D_CONV; ++j) {
    int ll = l - (D_CONV - 1) + j;
    if (ll >= 0)
      acc += cw[d * D_CONV + j] * xz[(size_t)(b * SEQ + ll) * (2 * D_INNER) + d];
  }
  float sv = acc / (1.f + __expf(-acc));
  xf[idx] = sv;
  xh[idx] = (_Float16)sv;
}

// dt-rank slice of x_dbl -> f16, zero-padded to 32 columns for dt_proj GEMM
__global__ void dtin_kernel(const float* __restrict__ xdbl,
                            _Float16* __restrict__ dtin, int n) {
  int idx = blockIdx.x * blockDim.x + threadIdx.x;
  if (idx >= n) return;
  int r = idx >> 5, c = idx & 31;
  dtin[idx] = (c < DT_RANK) ? (_Float16)xdbl[(size_t)r * XDBL_W + c]
                            : (_Float16)0.f;
}

// softplus(dt + bias), in place
__global__ void softplus_kernel(float* __restrict__ dt,
                                const float* __restrict__ dtb, int n) {
  int idx = blockIdx.x * blockDim.x + threadIdx.x;
  if (idx >= n) return;
  float v = dt[idx] + dtb[idx & (D_INNER - 1)];
  dt[idx] = (v > 20.f) ? v : log1pf(__expf(v));
}

// ---------------------------------------------------------------------------
// Selective scan: one lane per (b,d) channel, 16 states in registers.
// B/C (shared across d) staged through double-buffered LDS, 1 barrier/step.
// ---------------------------------------------------------------------------
__global__ __launch_bounds__(128) void scan_kernel(
    const float* __restrict__ dt, const float* __restrict__ xs,
    const float* __restrict__ xdbl, const float* __restrict__ Alog,
    float* __restrict__ y) {
  const int b = blockIdx.x;
  const int d = blockIdx.y * 128 + threadIdx.x;
  float A[D_STATE], h[D_STATE];
  #pragma unroll
  for (int s = 0; s < D_STATE; ++s) {
    A[s] = -__expf(Alog[(size_t)d * D_STATE + s]);
    h[s] = 0.f;
  }
  __shared__ float BC[2][2 * D_STATE];
  for (int l = 0; l < SEQ; ++l) {
    const int buf = l & 1;
    if (threadIdx.x < 2 * D_STATE)
      BC[buf][threadIdx.x] =
          xdbl[(size_t)(b * SEQ + l) * XDBL_W + DT_RANK + threadIdx.x];
    __syncthreads();
    const size_t o = (size_t)(b * SEQ + l) * D_INNER + d;
    const float dtv = dt[o];
    const float dx  = dtv * xs[o];
    float acc = 0.f;
    #pragma unroll
    for (int s = 0; s < D_STATE; ++s) {
      h[s] = __expf(dtv * A[s]) * h[s] + dx * BC[buf][s];
      acc += h[s] * BC[buf][D_STATE + s];
    }
    y[o] = acc;
  }
}

// y = (y_scan + x*Dp) * silu(z) -> f16 for out_proj
__global__ void ygate_kernel(const float* __restrict__ ys,
                             const float* __restrict__ xf,
                             const float* __restrict__ Dp,
                             const float* __restrict__ xz,
                             _Float16* __restrict__ yh, int n) {
  int idx = blockIdx.x * blockDim.x + threadIdx.x;
  if (idx >= n) return;
  int d = idx & (D_INNER - 1);
  int r = idx >> 9;
  float yv = ys[idx] + xf[idx] * Dp[d];
  float z = xz[(size_t)r * (2 * D_INNER) + D_INNER + d];
  yv *= z / (1.f + __expf(-z));
  yh[idx] = (_Float16)yv;
}

// hidden = hf + flip(hb); residual = res_f + flip(res_b)
__global__ void combine_kernel(const float* __restrict__ hf,
                               const float* __restrict__ hb,
                               const float* __restrict__ rf,
                               const float* __restrict__ rb,
                               float* __restrict__ hid,
                               float* __restrict__ res, int n) {
  int idx = blockIdx.x * blockDim.x + threadIdx.x;
  if (idx >= n) return;
  int d = idx & (D_MODEL - 1);
  int r = idx >> 8;
  int b = r / SEQ, l = r % SEQ;
  size_t fidx = (size_t)(b * SEQ + (SEQ - 1 - l)) * D_MODEL + d;
  hid[idx] = hf[idx] + hb[fidx];
  res[idx] = rf[idx] + rb[fidx];
}

// split + (B,L,D)->(B,D,L) permute for both modalities
__global__ void unpack_kernel(const float* __restrict__ nf,
                              float* __restrict__ out, int n) {
  int idx = blockIdx.x * blockDim.x + threadIdx.x;
  if (idx >= n) return;
  int l = idx % L1;
  int t = idx / L1;
  int d = t % D_MODEL;
  int b = t / D_MODEL;
  out[idx]     = nf[(size_t)(b * SEQ + l) * D_MODEL + d];
  out[n + idx] = nf[(size_t)(b * SEQ + L1 + l) * D_MODEL + d];
}

// ---------------------------------------------------------------------------
extern "C" void kernel_launch(void* const* d_in, const int* in_sizes, int n_in,
                              void* d_out, int out_size, void* d_ws, size_t ws_size,
                              hipStream_t stream) {
  (void)in_sizes; (void)n_in; (void)out_size; (void)ws_size;
  const float* rgb       = (const float*)d_in[0];
  const float* ir        = (const float*)d_in[1];
  const float* pos       = (const float*)d_in[2];
  const float* in_proj_w = (const float*)d_in[3];
  const float* conv_w    = (const float*)d_in[4];
  const float* conv_b    = (const float*)d_in[5];
  const float* x_proj_w  = (const float*)d_in[6];
  const float* dt_proj_w = (const float*)d_in[7];
  const float* dt_proj_b = (const float*)d_in[8];
  const float* A_log     = (const float*)d_in[9];
  const float* Dp        = (const float*)d_in[10];
  const float* out_proj_w= (const float*)d_in[11];
  const float* ln_w      = (const float*)d_in[12];
  const float* ln_b      = (const float*)d_in[13];
  const float* normf_w   = (const float*)d_in[14];
  const float* normf_b   = (const float*)d_in[15];

  char* base = (char*)d_ws;
  size_t off = 0;
  auto alloc = [&](size_t bytes) -> void* {
    void* p = base + off;
    off = (off + bytes + 255) & ~(size_t)255;
    return p;
  };
  const int M = M_ROWS;
  float*    hidden = (float*)   alloc((size_t)M * D_MODEL * 4);
  float*    resid  = (float*)   alloc((size_t)M * D_MODEL * 4);
  float*    resF   = (float*)   alloc((size_t)M * D_MODEL * 4);
  float*    resB   = (float*)   alloc((size_t)M * D_MODEL * 4);
  float*    hF     = (float*)   alloc((size_t)M * D_MODEL * 4);
  float*    hB     = (float*)   alloc((size_t)M * D_MODEL * 4);
  _Float16* normH  = (_Float16*)alloc((size_t)M * D_MODEL * 2);
  float*    xz     = (float*)   alloc((size_t)M * 2 * D_INNER * 4);
  float*    xsF    = (float*)   alloc((size_t)M * D_INNER * 4);
  _Float16* xsH    = (_Float16*)alloc((size_t)M * D_INNER * 2);
  float*    xdbl   = (float*)   alloc((size_t)M * XDBL_W * 4);
  _Float16* dtinH  = (_Float16*)alloc((size_t)M * DT_PAD * 2);
  float*    dtbuf  = (float*)   alloc((size_t)M * D_INNER * 4);
  float*    yscan  = (float*)   alloc((size_t)M * D_INNER * 4);
  _Float16* yH     = (_Float16*)alloc((size_t)M * D_INNER * 2);
  _Float16* inW    = (_Float16*)alloc((size_t)N_LAYERS * 2 * D_INNER * D_MODEL * 2);
  _Float16* xpW    = (_Float16*)alloc((size_t)N_LAYERS * XDBL_W * D_INNER * 2);
  _Float16* dtW    = (_Float16*)alloc((size_t)N_LAYERS * D_INNER * DT_PAD * 2);
  _Float16* outW   = (_Float16*)alloc((size_t)N_LAYERS * D_MODEL * D_INNER * 2);

  // Weight conversions (all layers at once); dt_proj_w padded 16 -> 32 K.
  {
    int n1 = N_LAYERS * 2 * D_INNER * D_MODEL;
    int n2 = N_LAYERS * XDBL_W * D_INNER;
    int n3 = N_LAYERS * D_INNER * DT_PAD;
    int n4 = N_LAYERS * D_MODEL * D_INNER;
    cvt_f16_kernel<<<(n1 + 255) / 256, 256, 0, stream>>>(in_proj_w, inW, n1);
    cvt_f16_kernel<<<(n2 + 255) / 256, 256, 0, stream>>>(x_proj_w, xpW, n2);
    cvt_pad_kernel<<<(n3 + 255) / 256, 256, 0, stream>>>(dt_proj_w, dtW, n3);
    cvt_f16_kernel<<<(n4 + 255) / 256, 256, 0, stream>>>(out_proj_w, outW, n4);
  }

  pack_kernel<<<(M * D_MODEL) / 256, 256, 0, stream>>>(rgb, ir, pos, hidden,
                                                       M * D_MODEL);

  auto gemm = [&](const _Float16* Aa, const _Float16* Ww, float* Cc,
                  int N, int K) {
    dim3 g((N / 16 + 3) / 4, M_ROWS / 64);
    if (K == D_MODEL)
      gemm_f16_wmma<D_MODEL><<<g, 128, 0, stream>>>(Aa, Ww, Cc, N);
    else if (K == D_INNER)
      gemm_f16_wmma<D_INNER><<<g, 128, 0, stream>>>(Aa, Ww, Cc, N);
    else
      gemm_f16_wmma<DT_PAD><<<g, 128, 0, stream>>>(Aa, Ww, Cc, N);
  };

  auto mamba = [&](int L, float* hout) {
    gemm(normH, inW + (size_t)L * 2 * D_INNER * D_MODEL, xz, 2 * D_INNER, D_MODEL);
    conv_silu_kernel<<<(M * D_INNER) / 256, 256, 0, stream>>>(
        xz, conv_w + (size_t)L * D_INNER * D_CONV, conv_b + (size_t)L * D_INNER,
        xsF, xsH, M * D_INNER);
    gemm(xsH, xpW + (size_t)L * XDBL_W * D_INNER, xdbl, XDBL_W, D_INNER);
    dtin_kernel<<<(M * DT_PAD) / 256, 256, 0, stream>>>(xdbl, dtinH, M * DT_PAD);
    gemm(dtinH, dtW + (size_t)L * D_INNER * DT_PAD, dtbuf, D_INNER, DT_PAD);
    softplus_kernel<<<(M * D_INNER) / 256, 256, 0, stream>>>(
        dtbuf, dt_proj_b + (size_t)L * D_INNER, M * D_INNER);
    scan_kernel<<<dim3(BSZ, D_INNER / 128), 128, 0, stream>>>(
        dtbuf, xsF, xdbl, A_log + (size_t)L * D_INNER * D_STATE, yscan);
    ygate_kernel<<<(M * D_INNER) / 256, 256, 0, stream>>>(
        yscan, xsF, Dp + (size_t)L * D_INNER, xz, yH, M * D_INNER);
    gemm(yH, outW + (size_t)L * D_MODEL * D_INNER, hout, D_MODEL, D_INNER);
  };

  const float* curRes = nullptr;
  for (int i = 0; i < N_LAYERS / 2; ++i) {
    const int lf = 2 * i, lb = 2 * i + 1;
    addnorm_kernel<<<M / 8, 256, 0, stream>>>(
        hidden, curRes, ln_w + (size_t)lf * D_MODEL, ln_b + (size_t)lf * D_MODEL,
        resF, normH, nullptr, 0);
    mamba(lf, hF);
    addnorm_kernel<<<M / 8, 256, 0, stream>>>(
        hidden, curRes, ln_w + (size_t)lb * D_MODEL, ln_b + (size_t)lb * D_MODEL,
        resB, normH, nullptr, 1);
    mamba(lb, hB);
    combine_kernel<<<(M * D_MODEL) / 256, 256, 0, stream>>>(
        hF, hB, resF, resB, hidden, resid, M * D_MODEL);
    curRes = resid;
  }

  // Final: residual += hidden, LayerNorm -> f32 (reuse hF), then unpack.
  addnorm_kernel<<<M / 8, 256, 0, stream>>>(hidden, curRes, normf_w, normf_b,
                                            resF, nullptr, hF, 0);
  unpack_kernel<<<(BSZ * D_MODEL * L1) / 256, 256, 0, stream>>>(
      hF, (float*)d_out, BSZ * D_MODEL * L1);
}
